// CrossAttention_9698036154819
// MI455X (gfx1250) — matrix operand order repaired and checked by reading, not compiled, not run on hardware
//
#include <hip/hip_runtime.h>

#define BATCH 8
#define NQL   2048
#define NKL   2048
#define DIM   512
#define BC    32          // key block
#define KSTR  (DIM + 8)   // Kt row stride (1040B = 65*16B, conflict-free, 16B aligned)
#define VSTR  (BC + 8)    // Vt row stride (80B = 5*16B, conflict-free, 16B aligned)

typedef __bf16 bf16;
typedef __attribute__((ext_vector_type(16))) __bf16 v16bf;
typedef __attribute__((ext_vector_type(8)))  __bf16 v8bf;
typedef __attribute__((ext_vector_type(8)))  float  v8f;
typedef __attribute__((ext_vector_type(16))) float  v16f;

__device__ __forceinline__ v8f wmma_bf16(v16bf a, v16bf b, v8f c) {
  // D = A(16x32 bf16) * B(32x16 bf16) + C(16x16 f32)
  return __builtin_amdgcn_wmma_f32_16x16x32_bf16(false, a, false, b, (short)0, c,
                                                 false, false);
}

__device__ __forceinline__ v16bf cat8(v8bf lo, v8bf hi) {
  return __builtin_shufflevector(lo, hi, 0, 1, 2, 3, 4, 5, 6, 7,
                                         8, 9, 10, 11, 12, 13, 14, 15);
}

// ---------------------------------------------------------------------------
// Tensor Data Mover staging of one K tile (BC x DIM bf16, row-major) into LDS
// with 4-dword padding after each 256-dword row (matches Kt[BC][KSTR]).
// ---------------------------------------------------------------------------
#if __has_builtin(__builtin_amdgcn_tensor_load_to_lds) && \
    __has_builtin(__builtin_amdgcn_s_wait_tensorcnt)
#define HAVE_TDM 1
typedef unsigned int v4u __attribute__((ext_vector_type(4)));
typedef int          v4i __attribute__((ext_vector_type(4)));
typedef int          v8i __attribute__((ext_vector_type(8)));

__device__ __forceinline__ void tdm_load_k_tile(const bf16* gptr,
                                                unsigned lds_off) {
  unsigned long long ga = (unsigned long long)(uintptr_t)gptr;
  v4u g0;
  g0[0] = 1u;                                  // count=1 (valid user descriptor)
  g0[1] = lds_off;                             // lds_addr (bytes)
  g0[2] = (unsigned)(ga & 0xffffffffu);        // global_addr[31:0]
  g0[3] = (unsigned)((ga >> 32) & 0x1ffffffu)  // global_addr[56:32]
          | (2u << 30);                        // type=2 ("image")
  v8i g1;
  g1[0] = (1 << 16)            // data_size = 2 bytes
        | (1 << 20)            // pad_enable
        | (7 << 22)            // pad_interval: 256 dwords (one 512-bf16 row)
        | (3 << 25);           // pad_amount: 4 dwords (8 bf16) -> KSTR stride
  g1[1] = (int)(DIM << 16);    // tensor_dim0[15:0] in bits [63:48]
  g1[2] = (int)(NKL << 16);    // tensor_dim1[15:0] in bits [95:80]
  g1[3] = (int)(DIM << 16);    // tile_dim0 in bits [127:112]
  g1[4] = BC;                  // tile_dim1 in bits [143:128]
  g1[5] = DIM;                 // tensor_dim0_stride[31:0] (elements)
  g1[6] = 0;
  g1[7] = 0;
  v4i gz = {0, 0, 0, 0};
#if defined(__clang_major__) && __clang_major__ >= 23
  v8i gz8 = {0, 0, 0, 0, 0, 0, 0, 0};
  __builtin_amdgcn_tensor_load_to_lds(g0, g1, gz, gz, gz8, 0);
#else
  __builtin_amdgcn_tensor_load_to_lds(g0, g1, gz, gz, 0);
#endif
}
#endif

// ---------------------------------------------------------------------------
// Projection: Y[row][e] = sum_d X[row][d] * W[e][d]   (y = x @ W^T), bf16 out.
// grid: (B*N/128, DIM/64, 3); block: 256 (8 waves). Wave w -> 16 rows, 64 cols.
// ---------------------------------------------------------------------------
__global__ __launch_bounds__(256) void proj_kernel(
    const float* __restrict__ Xq, const float* __restrict__ Xk,
    const float* __restrict__ Xv,
    const float* __restrict__ Wq, const float* __restrict__ Wk,
    const float* __restrict__ Wv,
    bf16* __restrict__ Yq, bf16* __restrict__ Yk, bf16* __restrict__ Yv) {
  const int which = blockIdx.z;
  const float* X = (which == 0) ? Xq : (which == 1) ? Xk : Xv;
  const float* W = (which == 0) ? Wq : (which == 1) ? Wk : Wv;
  bf16* Y        = (which == 0) ? Yq : (which == 1) ? Yk : Yv;

  const int lane = threadIdx.x & 31;
  const int wave = threadIdx.x >> 5;
  const int hf   = lane >> 4;   // lane half: selects K sub-range of A/B layouts
  const int l16  = lane & 15;

  const int row0 = blockIdx.x * 128 + wave * 16;
  const int col0 = blockIdx.y * 64;

  v8f acc[4] = {};
  const float* xrow = X + (size_t)(row0 + l16) * DIM;

#pragma unroll 2
  for (int c = 0; c < DIM; c += 32) {
    // A tile (16x32): lane row = l16; K = hf*16+0..15
    v8f xlo = *(const v8f*)(xrow + c + hf * 16);
    v8f xhi = *(const v8f*)(xrow + c + hf * 16 + 8);
    v16bf a;
#pragma unroll
    for (int i = 0; i < 8; ++i) { a[i] = (bf16)xlo[i]; a[i + 8] = (bf16)xhi[i]; }
#pragma unroll
    for (int j = 0; j < 4; ++j) {
      // B tile (32x16): lane col n = l16 (= output e), K = c + hf*16 + 0..15
      const float* wp = W + (size_t)(col0 + j * 16 + l16) * DIM + c + hf * 16;
      v16f wv = *(const v16f*)wp;
      v16bf b;
#pragma unroll
      for (int i = 0; i < 16; ++i) b[i] = (bf16)wv[i];
      acc[j] = wmma_bf16(a, b, acc[j]);
    }
  }
  // C layout: reg r -> row r (lanes 0-15) / r+8 (lanes 16-31); col = l16
#pragma unroll
  for (int j = 0; j < 4; ++j)
#pragma unroll
    for (int r = 0; r < 8; ++r)
      Y[(size_t)(row0 + r + hf * 8) * DIM + col0 + j * 16 + l16] = (bf16)acc[j][r];
}

// ---------------------------------------------------------------------------
// Flash attention, no 1/sqrt(d) scale (matches reference).
// grid: B*NQ/128; block 256 (8 waves); wave owns 16 query rows x full D=512.
// K tile staged by the Tensor Data Mover; V tile transposed into LDS manually.
// ---------------------------------------------------------------------------
__global__ __launch_bounds__(256) void attn_kernel(
    const bf16* __restrict__ Q, const bf16* __restrict__ K,
    const bf16* __restrict__ V, float* __restrict__ O) {
  __shared__ bf16 Kt[BC][KSTR];       // K block, row-major        (~33.3 KB)
  __shared__ bf16 Vt[DIM][VSTR];      // V block, transposed       (~40 KB)
  __shared__ bf16 Pw[8][16][BC];      // per-wave P re-layout      (8 KB)

  const int lane = threadIdx.x & 31;
  const int wave = threadIdx.x >> 5;
  const int hf   = lane >> 4;
  const int l16  = lane & 15;

  const int wg_per_b = NQL / 128;
  const int b        = blockIdx.x / wg_per_b;
  const int qrow0    = (blockIdx.x % wg_per_b) * 128 + wave * 16;

  const bf16* Qb = Q + ((size_t)b * NQL + qrow0) * DIM;
  const bf16* Kb = K + (size_t)b * NKL * DIM;
  const bf16* Vb = V + (size_t)b * NKL * DIM;

  // Preload the wave's 16 Q rows in A-matrix layout: aq[c] covers K=c*32..+31
  v16bf aq[16];
#pragma unroll
  for (int c = 0; c < 16; ++c) {
    const bf16* qp = Qb + (size_t)l16 * DIM + c * 32 + hf * 16;
    aq[c] = cat8(*(const v8bf*)qp, *(const v8bf*)(qp + 8));
  }

  v8f acc[32] = {};                   // 16 rows x 512 cols f32
  float m8[8], l8[8];
#pragma unroll
  for (int r = 0; r < 8; ++r) { m8[r] = -3.0e30f; l8[r] = 0.0f; }

#if defined(HAVE_TDM)
  const unsigned kt_lds = (unsigned)(uintptr_t)&Kt[0][0];
#endif

#pragma unroll 1
  for (int kb = 0; kb < NKL; kb += BC) {
    __syncthreads();                  // prior reads of Kt/Vt complete
#if defined(HAVE_TDM)
    if (wave == 0)                    // one DMA per workgroup (EXEC ignored)
      tdm_load_k_tile(Kb + (size_t)kb * DIM, kt_lds);
#else
#pragma unroll
    for (int i = 0; i < 8; ++i) {
      int idx = threadIdx.x + i * 256;
      int row = idx >> 6;
      int c8  = (idx & 63) * 8;
      *(v8bf*)&Kt[row][c8] = *(const v8bf*)(Kb + (size_t)(kb + row) * DIM + c8);
    }
#endif
    // ---- V tile: cooperative transpose into Vt (overlaps with TDM) ----
#pragma unroll
    for (int i = 0; i < 8; ++i) {
      int idx = threadIdx.x + i * 256;       // 2048 chunks of 8 bf16
      int row = idx >> 6;                    // 64 chunks per 512-wide row
      int c8  = (idx & 63) * 8;
      v8bf vv = *(const v8bf*)(Vb + (size_t)(kb + row) * DIM + c8);
#pragma unroll
      for (int e = 0; e < 8; ++e) Vt[c8 + e][row] = vv[e];
    }
    if (kb + BC < NKL) {                     // hint next block into caches
      __builtin_prefetch(Kb + (size_t)(kb + BC) * DIM, 0, 1);
      __builtin_prefetch(Vb + (size_t)(kb + BC) * DIM, 0, 1);
    }
#if defined(HAVE_TDM)
    if (wave == 0) __builtin_amdgcn_s_wait_tensorcnt(0);
#endif
    __syncthreads();

    // ---- S = Q * K^T  (16 x 32, two 16x16 tiles) ----
    v8f s0 = {}, s1 = {};
#pragma unroll
    for (int c = 0; c < 16; ++c) {
      // B tile: lane col = key, K = d (contiguous along Kt row)
      const bf16* k0 = &Kt[l16][c * 32 + hf * 16];        // keys 0..15
      const bf16* k1 = &Kt[l16 + 16][c * 32 + hf * 16];   // keys 16..31
      v16bf b0 = cat8(*(const v8bf*)k0, *(const v8bf*)(k0 + 8));
      v16bf b1 = cat8(*(const v8bf*)k1, *(const v8bf*)(k1 + 8));
      s0 = wmma_bf16(aq[c], b0, s0);
      s1 = wmma_bf16(aq[c], b1, s1);
    }

    // ---- online softmax (per-row over 16-lane halves) ----
    float mnew[8], scl[8];
#pragma unroll
    for (int r = 0; r < 8; ++r) {
      float mx = fmaxf(s0[r], s1[r]);
      mx = fmaxf(mx, __shfl_xor(mx, 1, 32));
      mx = fmaxf(mx, __shfl_xor(mx, 2, 32));
      mx = fmaxf(mx, __shfl_xor(mx, 4, 32));
      mx = fmaxf(mx, __shfl_xor(mx, 8, 32));
      mnew[r] = fmaxf(m8[r], mx);
      scl[r]  = __expf(m8[r] - mnew[r]);
      m8[r]   = mnew[r];
    }
#pragma unroll
    for (int r = 0; r < 8; ++r) {
      float p0 = __expf(s0[r] - mnew[r]);
      float p1 = __expf(s1[r] - mnew[r]);
      float rs = p0 + p1;
      rs += __shfl_xor(rs, 1, 32);
      rs += __shfl_xor(rs, 2, 32);
      rs += __shfl_xor(rs, 4, 32);
      rs += __shfl_xor(rs, 8, 32);
      l8[r] = l8[r] * scl[r] + rs;
      Pw[wave][r + hf * 8][l16]      = (bf16)p0;   // C layout -> LDS
      Pw[wave][r + hf * 8][l16 + 16] = (bf16)p1;
    }
#pragma unroll
    for (int t = 0; t < 32; ++t)
#pragma unroll
      for (int r = 0; r < 8; ++r) acc[t][r] *= scl[r];

    // reload P in A-matrix layout (same-wave DS ops are in-order)
    const bf16* pp = &Pw[wave][l16][hf * 16];
    v16bf pa = cat8(*(const v8bf*)pp, *(const v8bf*)(pp + 8));

    // ---- O += P * V : B tile from transposed V (contiguous along Vt row) ----
#pragma unroll
    for (int t = 0; t < 32; ++t) {
      const bf16* vp = &Vt[t * 16 + l16][hf * 16];
      v16bf bv = cat8(*(const v8bf*)vp, *(const v8bf*)(vp + 8));
      acc[t] = wmma_bf16(pa, bv, acc[t]);
    }
  }

  // ---- epilogue: O / l ----
  float inv[8];
#pragma unroll
  for (int r = 0; r < 8; ++r) inv[r] = 1.0f / l8[r];
  float* Ob = O + ((size_t)b * NQL + qrow0) * DIM;
#pragma unroll
  for (int t = 0; t < 32; ++t)
#pragma unroll
    for (int r = 0; r < 8; ++r)
      Ob[(size_t)(r + hf * 8) * DIM + t * 16 + l16] = acc[t][r] * inv[r];
}

// ---------------------------------------------------------------------------
extern "C" void kernel_launch(void* const* d_in, const int* in_sizes, int n_in,
                              void* d_out, int out_size, void* d_ws,
                              size_t ws_size, hipStream_t stream) {
  (void)in_sizes; (void)n_in; (void)out_size; (void)ws_size;
  const float* query = (const float*)d_in[0];
  const float* key   = (const float*)d_in[1];
  const float* value = (const float*)d_in[2];
  const float* Wq    = (const float*)d_in[3];
  const float* Wk    = (const float*)d_in[4];
  const float* Wv    = (const float*)d_in[5];
  float* out = (float*)d_out;

  const size_t tensor_elems = (size_t)BATCH * NQL * DIM;
  bf16* Qp = (bf16*)d_ws;
  bf16* Kp = Qp + tensor_elems;
  bf16* Vp = Kp + tensor_elems;

  dim3 gproj(BATCH * NQL / 128, DIM / 64, 3);
  proj_kernel<<<gproj, dim3(256), 0, stream>>>(query, key, value, Wq, Wk, Wv,
                                               Qp, Kp, Vp);

  dim3 gattn(BATCH * NQL / 128);
  attn_kernel<<<gattn, dim3(256), 0, stream>>>(Qp, Kp, Vp, out);
}
